// DBFLinear_NAIVE_30124900614509
// MI455X (gfx1250) — matrix operand order repaired and checked by reading, not compiled, not run on hardware
//
#include <hip/hip_runtime.h>

// CDNA5 / gfx1250, wave32.
typedef __attribute__((ext_vector_type(16))) _Float16 v16h;
typedef __attribute__((ext_vector_type(8)))  _Float16 v8h;
typedef __attribute__((ext_vector_type(8)))  float    v8f;

// Block = 256 threads = 8 waves. Block tile 128(M) x 128(N), K-step 32.
// Waves 4(M) x 2(N): each wave owns 32x64 = 2x4 WMMA 16x16 accumulators.
#define BM 128
#define BN 128
#define BK 32
#define LDA 48   // LDS row stride in halves (32 + 16 pad, keeps 16B alignment)

// ---------------------------------------------------------------------------
// Kernel 1: unpack sign bits -> (+/- rowscale) fp16 matrix, one byte/thread.
// bp holds one byte per int32 element, MSB-first; bit of W[n,k] = n*K + k.
// ---------------------------------------------------------------------------
__global__ __launch_bounds__(256)
void dbf_decode(const int* __restrict__ bp,
                const _Float16* __restrict__ rowscale,  // may be null
                _Float16* __restrict__ Wd,
                int rowBytes,        // K/8
                long totalBytes)     // N*K/8
{
    const long i = (long)blockIdx.x * 256 + threadIdx.x;
    if (i >= totalBytes) return;
    const unsigned byte = (unsigned)bp[i] & 0xFFu;
    _Float16 sp = (_Float16)1.0f;
    if (rowscale != nullptr) sp = rowscale[(int)(i / rowBytes)];
    const _Float16 sn = -sp;
    v8h o;
#pragma unroll
    for (int j = 0; j < 8; ++j)
        o[j] = ((byte >> (7 - j)) & 1u) ? sp : sn;
    *(v8h*)(Wd + (i << 3)) = o;
}

// ---------------------------------------------------------------------------
// Kernel 2: C[M,N] = (A[M,K] * colscale[K]) @ Wd[N,K]^T (+ bias[N]).
// Ping-pong double-buffered LDS. B tiles via CDNA5 async global->LDS DMA
// (ASYNCcnt); A tiles via load->pk_mul->ds_store (colscale fuse). The async
// and global-load latency of tile i+1 hides behind the 8 WMMAs of tile i.
// ---------------------------------------------------------------------------
template <bool HAS_BIAS>
__global__ __launch_bounds__(256)
void dbf_gemm(const _Float16* __restrict__ A,
              const _Float16* __restrict__ colscale,
              const _Float16* __restrict__ Bw,    // decoded (N,K) fp16
              const _Float16* __restrict__ bias,
              _Float16*       __restrict__ C,
              int M, int N, int K)
{
    __shared__ _Float16 sA[2][BM * LDA];  // 2 x 12 KB
    __shared__ _Float16 sB[2][BN * LDA];  // 2 x 12 KB

    const int tid  = threadIdx.x;
    const int lane = tid & 31;
    const int wid  = tid >> 5;
    const int wm   = wid & 3;   // wave M index 0..3
    const int wn   = wid >> 2;  // wave N index 0..1

    const int blockN0 = blockIdx.x * BN;
    const int blockM0 = blockIdx.y * BM;

    const int n_l = lane & 15;  // fragment column / row-within-16
    const int k_h = lane >> 4;  // which K-half this lane holds

    // A staging: 256 threads x 16 halves = 128x32 tile.
    const int arow = tid >> 1;
    const int aseg = (tid & 1) * 16;
    const _Float16* const Arow = A + (size_t)(blockM0 + arow) * (size_t)K + aseg;

    // Issue the async B-tile DMA for K-offset kk into buffer b (2 x b128/thread).
    auto issueB = [&](int kk, int b) {
#pragma unroll
        for (int j = 0; j < 2; ++j) {
            const int id   = tid + 256 * j;        // 0..511
            const int brow = id >> 2;              // 0..127
            const int bseg = (id & 3) * 8;         // halves: 0,8,16,24
            const _Float16* gsrc =
                Bw + (size_t)(blockN0 + brow) * (size_t)K + kk + bseg;
            const unsigned           ldsa =
                (unsigned)(uintptr_t)(&sB[b][brow * LDA + bseg]);
            const unsigned long long ga = (unsigned long long)(uintptr_t)gsrc;
            asm volatile("global_load_async_to_lds_b128 %0, %1, off"
                         :: "v"(ldsa), "v"(ga)
                         : "memory");
        }
    };

    v8f acc[2][4] = {};

    // ---- Prologue: stage tile 0 into buffer 0 ----
    issueB(0, 0);
    {
        v8h a0 = *(const v8h*)(Arow);
        v8h a1 = *(const v8h*)(Arow + 8);
        v8h c0 = *(const v8h*)(colscale + aseg);
        v8h c1 = *(const v8h*)(colscale + aseg + 8);
        a0 *= c0;
        a1 *= c1;
        _Float16* dst = &sA[0][arow * LDA + aseg];
        *(v8h*)(dst)     = a0;
        *(v8h*)(dst + 8) = a1;
    }
    asm volatile("s_wait_asynccnt 0x0" ::: "memory");
    __syncthreads();

    int buf = 0;
    for (int k0 = 0; k0 < K; k0 += BK) {
        const int kn = (k0 + BK < K) ? (k0 + BK) : 0;  // wrap: harmless prefetch
        const int nb = buf ^ 1;

        // 1) Kick off next tile's B DMA (writes nxt buffer only).
        issueB(kn, nb);

        // 2) Start next tile's A global loads (latency hidden by WMMAs below).
        v8h a0 = *(const v8h*)(Arow + kn);
        v8h a1 = *(const v8h*)(Arow + kn + 8);
        v8h c0 = *(const v8h*)(colscale + kn + aseg);
        v8h c1 = *(const v8h*)(colscale + kn + aseg + 8);
        __builtin_prefetch(Arow + kn + BK, 0, 1);  // global_prefetch_b8

        // 3) Fragments from current buffer.
        // A (16-bit 16x32 layout): lane {k_h, m}; VGPR0-3: K=8h+0..7; VGPR4-7: K=16+8h+0..7.
        v16h afrag[2];
#pragma unroll
        for (int mi = 0; mi < 2; ++mi) {
            const _Float16* ab = &sA[buf][(wm * 32 + mi * 16 + n_l) * LDA + 8 * k_h];
            v8h lo = *(const v8h*)(ab);
            v8h hi = *(const v8h*)(ab + 16);
#pragma unroll
            for (int j = 0; j < 8; ++j) {
                afrag[mi][j]     = lo[j];
                afrag[mi][8 + j] = hi[j];
            }
        }
        // B: lane {k_h, n}; VGPR0-3: K=8h+0..7; VGPR4-7: K=16+8h+0..7.
        v16h bfrag[4];
#pragma unroll
        for (int ni = 0; ni < 4; ++ni) {
            const _Float16* bb = &sB[buf][(wn * 64 + ni * 16 + n_l) * LDA + 8 * k_h];
            v8h lo = *(const v8h*)(bb);
            v8h hi = *(const v8h*)(bb + 16);
#pragma unroll
            for (int j = 0; j < 8; ++j) {
                bfrag[ni][j]     = lo[j];
                bfrag[ni][8 + j] = hi[j];
            }
        }

        // 4) 2x4 WMMA block (fp16 in, fp32 acc).
#pragma unroll
        for (int mi = 0; mi < 2; ++mi) {
#pragma unroll
            for (int ni = 0; ni < 4; ++ni) {
                acc[mi][ni] = __builtin_amdgcn_wmma_f32_16x16x32_f16(
                    false, afrag[mi], false, bfrag[ni],
                    (short)0, acc[mi][ni], false, false);
            }
        }

        // 5) Finish staging next A tile (scale fuse + LDS store to nxt buffer).
        a0 *= c0;
        a1 *= c1;
        _Float16* dst = &sA[nb][arow * LDA + aseg];
        *(v8h*)(dst)     = a0;
        *(v8h*)(dst + 8) = a1;

        // 6) This iteration's 2 async ops are the only outstanding ones.
        asm volatile("s_wait_asynccnt 0x0" ::: "memory");
        __syncthreads();
        buf = nb;
    }

    // ---- Epilogue: optional bias, convert to fp16, store ----
    // 16x16 f32 C/D layout: lane {hi,n}: VGPR r holds M = r + 8*hi, N = n.
#pragma unroll
    for (int mi = 0; mi < 2; ++mi) {
#pragma unroll
        for (int ni = 0; ni < 4; ++ni) {
            const int gn = blockN0 + wn * 64 + ni * 16 + n_l;
            float bv = 0.0f;
            if constexpr (HAS_BIAS) bv = (float)bias[gn];
            const int gm0 = blockM0 + wm * 32 + mi * 16 + 8 * k_h;
#pragma unroll
            for (int r = 0; r < 8; ++r) {
                C[(size_t)(gm0 + r) * (size_t)N + gn] =
                    (_Float16)(acc[mi][ni][r] + bv);
            }
        }
    }
}

extern "C" void kernel_launch(void* const* d_in, const int* in_sizes, int n_in,
                              void* d_out, int out_size, void* d_ws, size_t ws_size,
                              hipStream_t stream)
{
    const _Float16* x    = (const _Float16*)d_in[0];  // (B, IN) fp16
    const _Float16* s0   = (const _Float16*)d_in[1];  // (IN,)
    const int*      bp1  = (const int*)d_in[2];       // packed W1 (MID, IN)
    const _Float16* s2   = (const _Float16*)d_in[3];  // (MID,)
    const int*      bp3  = (const int*)d_in[4];       // packed W3 (OUT, MID)
    const _Float16* s4   = (const _Float16*)d_in[5];  // (OUT,)
    const _Float16* bias = (const _Float16*)d_in[6];  // (OUT,)

    const int IN  = in_sizes[1];
    const int MID = in_sizes[3];
    const int OUT = in_sizes[5];
    const int B   = in_sizes[0] / IN;

    char* ws = (char*)d_ws;
    _Float16* H  = (_Float16*)ws;                          // (B, MID): 64 MB
    _Float16* Wd = (_Float16*)(ws + (size_t)B * MID * 2);  // 32 MB, reused (L2-resident)

    _Float16* out = (_Float16*)d_out;  // (B, OUT) fp16

    dim3 blk(256, 1, 1);

    // Layer 1: decode W1 (unscaled), H = (x * s0) @ W1^T
    const long nb1 = (long)MID * IN / 8;
    dbf_decode<<<dim3((unsigned)((nb1 + 255) / 256)), blk, 0, stream>>>(
        bp1, nullptr, Wd, IN / 8, nb1);
    dim3 g1(MID / BN, B / BM, 1);
    dbf_gemm<false><<<g1, blk, 0, stream>>>(x, s0, Wd, nullptr, H, B, MID, IN);

    // Layer 2: decode W3 scaled by s4, Y = ((H * s2) @ (s4*W3)^T) + bias
    const long nb2 = (long)OUT * MID / 8;
    dbf_decode<<<dim3((unsigned)((nb2 + 255) / 256)), blk, 0, stream>>>(
        bp3, s4, Wd, MID / 8, nb2);
    dim3 g2(OUT / BN, B / BM, 1);
    dbf_gemm<true><<<g2, blk, 0, stream>>>(H, s2, Wd, bias, out, B, OUT, MID);
}
